// AttentionDecoder_84430467105235
// MI455X (gfx1250) — matrix operand order repaired and checked by reading, not compiled, not run on hardware
//
#include <hip/hip_runtime.h>
#include <hip/hip_bf16.h>

// ---------------------------------------------------------------------------
// MI455X attention-decoder step.
// Bandwidth-bound (672 MB fp32 weights @ 23.3 TB/s ~= 29 us floor), so:
//  * all weight GEMVs read with b128 loads, split-K for occupancy,
//  * matrix math runs on v_wmma_f32_16x16x32_f16 using a 3-term split-f16
//    decomposition (Wh*xh + Wh*xl + Wl*xh) => ~fp32 accuracy, zero extra HBM,
//  * x-slices staged into LDS by the Tensor Data Mover (tensor_load_to_lds,
//    s_wait_tensorcnt) -- one DMA op per block instead of per-thread loads,
//  * deterministic two-stage reductions (no float atomics).
// ---------------------------------------------------------------------------

typedef _Float16 v16h __attribute__((ext_vector_type(16)));
typedef float    v8f  __attribute__((ext_vector_type(8)));
typedef unsigned int u32x4 __attribute__((ext_vector_type(4)));
typedef int          i32x8 __attribute__((ext_vector_type(8)));
typedef int          i32x4 __attribute__((ext_vector_type(4)));

__device__ __forceinline__ void split_push(v16h& vh, v16h& vl, const float4 f, const int i0) {
    const float fs[4] = {f.x, f.y, f.z, f.w};
#pragma unroll
    for (int t = 0; t < 4; ++t) {
        _Float16 h = (_Float16)fs[t];
        vh[i0 + t] = h;
        vl[i0 + t] = (_Float16)(fs[t] - (float)h);
    }
}

// out_r = dot(W[r, :], x) over K-slice [s*kslice, (s+1)*kslice), partials to P.
// One wave per 16 rows; A = 16x32 fp32 weight tile split to f16 hi/lo,
// B = x chunk broadcast across all 16 columns (lane=N, lane-group selects K half).
__global__ void __launch_bounds__(256)
gemv_wmma_kernel(const float* __restrict__ W, const float* __restrict__ x,
                 float* __restrict__ P, int rows, int K, int kslice)
{
    extern __shared__ float sx[];                      // kslice floats
    const int tid = threadIdx.x;
    const int s   = blockIdx.y;
    const int k0  = s * kslice;

    // --- TDM: one 1-row tile DMA (kslice fp32) global -> LDS, issued by wave 0.
    if (tid < 32) {
        const unsigned long long ga = (unsigned long long)(uintptr_t)(x + k0);
        const unsigned int ldsOff   = (unsigned int)(uintptr_t)(&sx[0]); // low 32 bits = LDS byte offset
        const unsigned int td0      = (unsigned int)kslice;              // elements (fp32)
        u32x4 g0;
        g0[0] = 1u;                                            // count=1, user descriptor
        g0[1] = ldsOff;                                        // lds_addr (bytes)
        g0[2] = (unsigned int)(ga & 0xffffffffu);              // global_addr[31:0]
        g0[3] = (unsigned int)((ga >> 32) & 0x1ffffffu)        // global_addr[56:32]
              | (2u << 30);                                    // type=2 ("image")
        i32x8 g1;
        g1[0] = (int)(2u << 16);                               // data_size=2 -> 4 bytes, mask=0
        g1[1] = (int)((td0 & 0xffffu) << 16);                  // tensor_dim0[15:0]
        g1[2] = (int)(((td0 >> 16) & 0xffffu) | (1u << 16));   // tensor_dim0[31:16], tensor_dim1=1
        g1[3] = (int)((td0 & 0xffffu) << 16);                  // tile_dim0 = kslice
        g1[4] = 1;                                             // tile_dim1 = 1 (tile_dim2 = 0)
        g1[5] = (int)td0;                                      // tensor_dim0_stride[31:0]
        g1[6] = 0;                                             // stride msbs / dim1 stride
        g1[7] = 0;
        const i32x4 z4 = {0, 0, 0, 0};                         // groups 2/3 unused (<=2D tile)
        const i32x8 z8 = {0, 0, 0, 0, 0, 0, 0, 0};
        __builtin_amdgcn_tensor_load_to_lds(g0, g1, z4, z4, z8, 0);
        __builtin_amdgcn_s_wait_tensorcnt(0);
    }
    __syncthreads();

    const int wave = tid >> 5;
    const int lane = tid & 31;
    const int hi   = lane >> 4;                        // K lane-group select
    const int r0   = (blockIdx.x * 8 + wave) * 16;
    const long rowBase = (long)(r0 + (lane & 15)) * K;

    v8f c = {0.f, 0.f, 0.f, 0.f, 0.f, 0.f, 0.f, 0.f};

#pragma unroll 2
    for (int kc = 0; kc < kslice; kc += 32) {
        // A tile: lane group 0 -> K {0..7,16..23}, group 1 -> K {8..15,24..31}
        const float* wp = W + rowBase + k0 + kc + hi * 8;
        __builtin_prefetch(wp + 32, 0, 1);             // global_prefetch next chunk
        const float4 a0 = *(const float4*)(wp);
        const float4 a1 = *(const float4*)(wp + 4);
        const float4 a2 = *(const float4*)(wp + 16);
        const float4 a3 = *(const float4*)(wp + 20);
        v16h ah, al;
        split_push(ah, al, a0, 0);
        split_push(ah, al, a1, 4);
        split_push(ah, al, a2, 8);
        split_push(ah, al, a3, 12);

        // B (column-broadcast): lane group 0 -> K {0..7,16..23}, group 1 -> K {8..15,24..31}
        const float* xp = &sx[kc + hi * 8];
        const float4 x0 = *(const float4*)(xp);
        const float4 x1 = *(const float4*)(xp + 4);
        const float4 x2 = *(const float4*)(xp + 16);
        const float4 x3 = *(const float4*)(xp + 20);
        v16h bh, bl;
        split_push(bh, bl, x0, 0);
        split_push(bh, bl, x1, 4);
        split_push(bh, bl, x2, 8);
        split_push(bh, bl, x3, 12);

        c = __builtin_amdgcn_wmma_f32_16x16x32_f16(false, ah, false, bh, (short)0, c, false, false);
        c = __builtin_amdgcn_wmma_f32_16x16x32_f16(false, ah, false, bl, (short)0, c, false, false);
        c = __builtin_amdgcn_wmma_f32_16x16x32_f16(false, al, false, bh, (short)0, c, false, false);
    }

    // All D columns equal: lane 0 carries rows r0..r0+7, lane 16 rows r0+8..r0+15.
    if ((lane & 15) == 0) {
        float* dst = P + (long)s * rows + r0 + hi * 8;
#pragma unroll
        for (int g = 0; g < 8; ++g) dst[g] = c[g];
    }
}

// emb gather + build x1 = [emb | h0], x2[0:4096] = emb
__global__ void __launch_bounds__(256)
prep_kernel(const int* __restrict__ idx, const float* __restrict__ embW,
            const float* __restrict__ hidden, float* __restrict__ x1,
            float* __restrict__ x2)
{
    const int j = blockIdx.x * 256 + threadIdx.x;       // 4096 threads
    const float e = embW[(long)idx[0] * 4096 + j];
    x1[j] = e;
    x1[4096 + j] = hidden[j];
    x2[j] = e;
}

// fused: logits = bias + sum_s P[s][:]; softmax -> out (2048)
__global__ void __launch_bounds__(256)
softmax_att_kernel(const float* __restrict__ P, const float* __restrict__ bias,
                   float* __restrict__ out, int S)
{
    __shared__ float sl[2048];
    __shared__ float red[256];
    const int tid = threadIdx.x;
    float lmax = -3.4e38f;
    for (int j = tid; j < 2048; j += 256) {
        float v = bias[j];
        for (int s = 0; s < S; ++s) v += P[(long)s * 2048 + j];
        sl[j] = v;
        lmax = fmaxf(lmax, v);
    }
    red[tid] = lmax; __syncthreads();
    for (int off = 128; off > 0; off >>= 1) {
        if (tid < off) red[tid] = fmaxf(red[tid], red[tid + off]);
        __syncthreads();
    }
    const float gmax = red[0]; __syncthreads();
    float lsum = 0.f;
    for (int j = tid; j < 2048; j += 256) {
        const float e = expf(sl[j] - gmax);
        sl[j] = e;
        lsum += e;
    }
    red[tid] = lsum; __syncthreads();
    for (int off = 128; off > 0; off >>= 1) {
        if (tid < off) red[tid] += red[tid + off];
        __syncthreads();
    }
    const float inv = 1.f / red[0]; __syncthreads();
    for (int j = tid; j < 2048; j += 256) out[j] = sl[j] * inv;
}

// atapplied partials: P[s][:] += w[k] * E[k][:]  (row-major E => fully coalesced)
__global__ void __launch_bounds__(256)
attn_apply_kernel(const float* __restrict__ w, const float* __restrict__ E,
                  float* __restrict__ P)
{
    __shared__ float sw[128];
    const int tid = threadIdx.x;
    const int j4  = blockIdx.x * 256 + tid;             // float4 column, [0,1024)
    const int k0  = blockIdx.y * 128;
    if (tid < 128) sw[tid] = w[k0 + tid];
    __syncthreads();
    float4 acc = {0.f, 0.f, 0.f, 0.f};
#pragma unroll 4
    for (int k = 0; k < 128; ++k) {
        const float wk  = sw[k];
        const float4 e = *(const float4*)(E + (long)(k0 + k) * 4096 + j4 * 4);
        acc.x += wk * e.x; acc.y += wk * e.y; acc.z += wk * e.z; acc.w += wk * e.w;
    }
    *(float4*)(P + (long)blockIdx.y * 4096 + j4 * 4) = acc;
}

// dst[j] = act(bias[j] + sum_s P[s*n + j])
__global__ void __launch_bounds__(256)
reduce_bias_act_kernel(const float* __restrict__ P, const float* __restrict__ bias,
                       float* __restrict__ dst, int n, int S, int act)
{
    const int j = blockIdx.x * 256 + threadIdx.x;
    if (j >= n) return;
    float v = bias ? bias[j] : 0.f;
    for (int s = 0; s < S; ++s) v += P[(long)s * n + j];
    if (act) v = fmaxf(v, 0.f);
    dst[j] = v;
}

// fused GRU gate math (reduces gi/gh partials inline)
__global__ void __launch_bounds__(256)
gru_gate_kernel(const float* __restrict__ Pgi, const float* __restrict__ Pgh,
                const float* __restrict__ b_ih, const float* __restrict__ b_hh,
                const float* __restrict__ h0, float* __restrict__ hnew, int S)
{
    const int j = blockIdx.x * 256 + threadIdx.x;       // 4096 threads
    float ir = b_ih[j], iz = b_ih[4096 + j], in_ = b_ih[8192 + j];
    float hr = b_hh[j], hz = b_hh[4096 + j], hn  = b_hh[8192 + j];
    for (int s = 0; s < S; ++s) {
        const float* pi = Pgi + (long)s * 12288;
        const float* ph = Pgh + (long)s * 12288;
        ir += pi[j]; iz += pi[4096 + j]; in_ += pi[8192 + j];
        hr += ph[j]; hz += ph[4096 + j]; hn  += ph[8192 + j];
    }
    const float r = 1.f / (1.f + expf(-(ir + hr)));
    const float z = 1.f / (1.f + expf(-(iz + hz)));
    const float n = tanhf(in_ + r * hn);
    hnew[j] = (1.f - z) * n + z * h0[j];
}

// fused: v = bias + sum_s partials; log_softmax -> out (4096)
__global__ void __launch_bounds__(256)
logsoftmax_kernel(const float* __restrict__ P, const float* __restrict__ bias,
                  float* __restrict__ out, int S)
{
    __shared__ float sl[4096];
    __shared__ float red[256];
    const int tid = threadIdx.x;
    float lmax = -3.4e38f;
    for (int j = tid; j < 4096; j += 256) {
        float v = bias[j];
        for (int s = 0; s < S; ++s) v += P[(long)s * 4096 + j];
        sl[j] = v;
        lmax = fmaxf(lmax, v);
    }
    red[tid] = lmax; __syncthreads();
    for (int off = 128; off > 0; off >>= 1) {
        if (tid < off) red[tid] = fmaxf(red[tid], red[tid + off]);
        __syncthreads();
    }
    const float gmax = red[0]; __syncthreads();
    float lsum = 0.f;
    for (int j = tid; j < 4096; j += 256) lsum += expf(sl[j] - gmax);
    red[tid] = lsum; __syncthreads();
    for (int off = 128; off > 0; off >>= 1) {
        if (tid < off) red[tid] += red[tid + off];
        __syncthreads();
    }
    const float lse = logf(red[0]); __syncthreads();
    for (int j = tid; j < 4096; j += 256) out[j] = sl[j] - gmax - lse;
}

extern "C" void kernel_launch(void* const* d_in, const int* in_sizes, int n_in,
                              void* d_out, int out_size, void* d_ws, size_t ws_size,
                              hipStream_t stream)
{
    (void)in_sizes; (void)n_in; (void)out_size; (void)ws_size;
    const int*   idx    = (const int*)  d_in[0];
    const float* hidden = (const float*)d_in[1];
    const float* encs   = (const float*)d_in[3];   // encoder_outputs [2048,4096]
    const float* embW   = (const float*)d_in[4];
    const float* attW   = (const float*)d_in[5];
    const float* attb   = (const float*)d_in[6];
    const float* atcW   = (const float*)d_in[7];
    const float* atcb   = (const float*)d_in[8];
    const float* Wih    = (const float*)d_in[9];
    const float* Whh    = (const float*)d_in[10];
    const float* bih    = (const float*)d_in[11];
    const float* bhh    = (const float*)d_in[12];
    const float* outW   = (const float*)d_in[13];
    const float* outb   = (const float*)d_in[14];

    float* out_ls  = (float*)d_out;        // log_softmax  [4096]
    float* out_h   = out_ls + 4096;        // h_new        [4096]
    float* out_att = out_h  + 4096;        // atweights    [2048]

    float* ws    = (float*)d_ws;           // ~868 KB used, regions reused
    float* x1    = ws;                     // 8192: [emb | h0]
    float* x2    = ws + 8192;              // 8192: [emb | atapplied]
    float* xrelu = ws + 16384;             // 4096
    float* R     = ws + 20480;             // up to 196608 floats of partials
    float* Pgi   = R;
    float* Pgh   = R + 98304;              // 8*12288

    const int KS = 512;                    // split-K slice
    const size_t shmem = KS * sizeof(float);

    prep_kernel<<<16, 256, 0, stream>>>(idx, embW, hidden, x1, x2);

    // attention logits: att_W [2048 x 8192]
    gemv_wmma_kernel<<<dim3(16, 16), 256, shmem, stream>>>(attW, x1, R, 2048, 8192, KS);
    softmax_att_kernel<<<1, 256, 0, stream>>>(R, attb, out_att, 16);

    // atapplied = atweights @ encoder_outputs  (coalesced column GEMV)
    attn_apply_kernel<<<dim3(4, 16), 256, 0, stream>>>(out_att, encs, R);
    reduce_bias_act_kernel<<<16, 256, 0, stream>>>(R, nullptr, x2 + 4096, 4096, 16, 0);

    // x = relu(comb @ atc_W.T + atc_b): atc_W [4096 x 8192]
    gemv_wmma_kernel<<<dim3(32, 16), 256, shmem, stream>>>(atcW, x2, R, 4096, 8192, KS);
    reduce_bias_act_kernel<<<16, 256, 0, stream>>>(R, atcb, xrelu, 4096, 16, 1);

    // GRU gates: W_ih/W_hh [12288 x 4096]
    gemv_wmma_kernel<<<dim3(96, 8), 256, shmem, stream>>>(Wih, xrelu,  Pgi, 12288, 4096, KS);
    gemv_wmma_kernel<<<dim3(96, 8), 256, shmem, stream>>>(Whh, hidden, Pgh, 12288, 4096, KS);
    gru_gate_kernel<<<16, 256, 0, stream>>>(Pgi, Pgh, bih, bhh, hidden, out_h, 8);

    // out = log_softmax(h_new @ out_W.T + out_b): out_W [4096 x 4096]
    gemv_wmma_kernel<<<dim3(32, 8), 256, shmem, stream>>>(outW, out_h, R, 4096, 4096, KS);
    logsoftmax_kernel<<<1, 256, 0, stream>>>(R, outb, out_ls, 8);
}
